// Agent_25546465476958
// MI455X (gfx1250) — matrix operand order repaired and checked by reading, not compiled, not run on hardware
//
#include <hip/hip_runtime.h>

typedef __attribute__((ext_vector_type(16))) _Float16 v16h;
typedef __attribute__((ext_vector_type(8)))  _Float16 v8h;
typedef __attribute__((ext_vector_type(8)))  float    v8f;

namespace {
constexpr int kB      = 256;
constexpr int kHID    = 512;
constexpr int kMEM    = 64;
constexpr int kL      = 65;       // MEM + 1
constexpr int kDIN    = 1024;     // EXPAND * HID
constexpr int kROWS   = kB * kL;  // 16640
constexpr int kNACT   = 18;
constexpr int kFCK    = 64 * 7 * 7; // 3136
}

// ---------------------------------------------------------------------------
// Fragment loaders (16-bit WMMA VGPR layouts, wave32)
//   A 16x32: lane<16 -> K {0..7,16..23}, lane>=16 -> K {8..15,24..31}
//   B 32x16: lane<16 -> K {0..7,16..23}, lane>=16 -> K {8..15,24..31} (same as A)
// Caller bakes the per-lane K offset into the pointer; loads become
// base-pointer + immediate-offset b128 pairs.
// ---------------------------------------------------------------------------
__device__ __forceinline__ v16h load_afrag(const _Float16* __restrict__ p)
{
    v8h lo = *(const v8h*)(p);
    v8h hi = *(const v8h*)(p + 16);
    return __builtin_shufflevector(lo, hi, 0,1,2,3,4,5,6,7,8,9,10,11,12,13,14,15);
}
__device__ __forceinline__ v16h load_bfrag(const _Float16* __restrict__ p)
{
    v8h lo = *(const v8h*)(p);
    v8h hi = *(const v8h*)(p + 16);
    return __builtin_shufflevector(lo, hi, 0,1,2,3,4,5,6,7,8,9,10,11,12,13,14,15);
}

// ---------------------------------------------------------------------------
// WMMA GEMM:  out[M,N] = act( A[M,K](f16) * W[N,K](f16)^T + bias )
// block = 128 threads (4 waves); wave tile = 64(M) x 32(N); 8 WMMAs / k-step.
// Register double-buffered fragments, k unrolled by 64; launch_bounds(128,1)
// so the RA keeps acc + both fragment buffers resident (no scratch spills).
// Requires K % 64 == 0 (true for 3136 / 512 / 1024).
// ---------------------------------------------------------------------------
__global__ __launch_bounds__(128, 1)
void gemm_wmma_f16(const _Float16* __restrict__ A,
                   const _Float16* __restrict__ W,
                   const float* __restrict__ bias,
                   float* __restrict__ outF,
                   _Float16* __restrict__ outH,
                   int M, int N, int K, int relu)
{
    const int lane   = threadIdx.x & 31;
    const int wave   = threadIdx.x >> 5;
    const int n0     = blockIdx.x * 128 + wave * 32;
    const int m0     = blockIdx.y * 64;
    if (n0 >= N || m0 >= M) return;

    const int l15    = lane & 15;
    const int hiHalf = (lane >> 4) & 1;
    const int aoff   = hiHalf * 8;    // A frag lane K-offset
    const int boff   = hiHalf * 8;    // B frag lane K-offset

    v8f zero = {};
    v8f acc[4][2];
#pragma unroll
    for (int i = 0; i < 4; ++i)
#pragma unroll
        for (int j = 0; j < 2; ++j) acc[i][j] = zero;

    const _Float16* aRow[4];
#pragma unroll
    for (int i = 0; i < 4; ++i)
        aRow[i] = A + (size_t)(m0 + i * 16 + l15) * K + aoff;
    const _Float16* bRow[2];
#pragma unroll
    for (int j = 0; j < 2; ++j)
        bRow[j] = W + (size_t)(n0 + j * 16 + l15) * K + boff;

    v16h a0[4], a1[4];
    v16h b0[2], b1[2];

    // prologue: buffer 0 <- frags(k=0)
#pragma unroll
    for (int i = 0; i < 4; ++i) a0[i] = load_afrag(aRow[i]);
#pragma unroll
    for (int j = 0; j < 2; ++j) b0[j] = load_bfrag(bRow[j]);

    int k = 0;
    // invariant at loop top: buf0 holds frags(k)
    for (; k + 64 < K; k += 64) {
        // issue loads for k+32 while WMMAs consume buf0
#pragma unroll
        for (int i = 0; i < 4; ++i) {
            a1[i] = load_afrag(aRow[i] + k + 32);
            __builtin_prefetch(aRow[i] + k + 96, 0, 1);
        }
#pragma unroll
        for (int j = 0; j < 2; ++j) {
            b1[j] = load_bfrag(bRow[j] + k + 32);
            __builtin_prefetch(bRow[j] + k + 96, 0, 1);
        }
#pragma unroll
        for (int i = 0; i < 4; ++i)
#pragma unroll
            for (int j = 0; j < 2; ++j)
                acc[i][j] = __builtin_amdgcn_wmma_f32_16x16x32_f16(
                    false, a0[i], false, b0[j], (short)0, acc[i][j], false, false);

        // issue loads for k+64 while WMMAs consume buf1
#pragma unroll
        for (int i = 0; i < 4; ++i) a0[i] = load_afrag(aRow[i] + k + 64);
#pragma unroll
        for (int j = 0; j < 2; ++j) b0[j] = load_bfrag(bRow[j] + k + 64);
#pragma unroll
        for (int i = 0; i < 4; ++i)
#pragma unroll
            for (int j = 0; j < 2; ++j)
                acc[i][j] = __builtin_amdgcn_wmma_f32_16x16x32_f16(
                    false, a1[i], false, b1[j], (short)0, acc[i][j], false, false);
    }

    // epilogue: steps k and k+32 (k == K-64 here)
#pragma unroll
    for (int i = 0; i < 4; ++i) a1[i] = load_afrag(aRow[i] + k + 32);
#pragma unroll
    for (int j = 0; j < 2; ++j) b1[j] = load_bfrag(bRow[j] + k + 32);
#pragma unroll
    for (int i = 0; i < 4; ++i)
#pragma unroll
        for (int j = 0; j < 2; ++j)
            acc[i][j] = __builtin_amdgcn_wmma_f32_16x16x32_f16(
                false, a0[i], false, b0[j], (short)0, acc[i][j], false, false);
#pragma unroll
    for (int i = 0; i < 4; ++i)
#pragma unroll
        for (int j = 0; j < 2; ++j)
            acc[i][j] = __builtin_amdgcn_wmma_f32_16x16x32_f16(
                false, a1[i], false, b1[j], (short)0, acc[i][j], false, false);

    // C/D layout: lane<16 -> N=lane, M=r ; lane>=16 -> N=lane-16, M=8+r
#pragma unroll
    for (int i = 0; i < 4; ++i) {
#pragma unroll
        for (int j = 0; j < 2; ++j) {
            const int ncol = n0 + j * 16 + l15;
            const float bval = bias ? bias[ncol] : 0.0f;
#pragma unroll
            for (int r = 0; r < 8; ++r) {
                const int mrow = m0 + i * 16 + hiHalf * 8 + r;
                float v = acc[i][j][r] + bval;
                if (relu) v = fmaxf(v, 0.0f);
                const size_t o = (size_t)mrow * N + ncol;
                if (outH) outH[o] = (_Float16)v;
                else      outF[o] = v;
            }
        }
    }
}

// ---------------------------------------------------------------------------
// Encoder convs (direct, VALU; small FLOP fraction)
// ---------------------------------------------------------------------------
__global__ void conv1_k(const float* __restrict__ x, const float* __restrict__ w,
                        const float* __restrict__ bias, float* __restrict__ out)
{   // (256,3,84,84) -> (256,32,20,20), k=8 s=4, fused /255 + ReLU
    int idx = blockIdx.x * blockDim.x + threadIdx.x;
    if (idx >= kB * 32 * 20 * 20) return;
    int ox = idx % 20, oy = (idx / 20) % 20, oc = (idx / 400) % 32, b = idx / (400 * 32);
    const float* xb = x + (size_t)b * 3 * 84 * 84;
    float acc = 0.0f;
    for (int ic = 0; ic < 3; ++ic) {
#pragma unroll
        for (int ky = 0; ky < 8; ++ky) {
            const float* xr = xb + ic * 7056 + (oy * 4 + ky) * 84 + ox * 4;
            const float* wr = w + oc * 192 + ic * 64 + ky * 8;
#pragma unroll
            for (int kx = 0; kx < 8; ++kx) acc += xr[kx] * wr[kx];
        }
    }
    out[idx] = fmaxf(acc * (1.0f / 255.0f) + bias[oc], 0.0f);
}

__global__ void conv2_k(const float* __restrict__ h1, const float* __restrict__ w,
                        const float* __restrict__ bias, float* __restrict__ out)
{   // (256,32,20,20) -> (256,64,9,9), k=4 s=2, ReLU
    int idx = blockIdx.x * blockDim.x + threadIdx.x;
    if (idx >= kB * 64 * 81) return;
    int ox = idx % 9, oy = (idx / 9) % 9, oc = (idx / 81) % 64, b = idx / (81 * 64);
    const float* xb = h1 + (size_t)b * 32 * 400;
    float acc = 0.0f;
    for (int ic = 0; ic < 32; ++ic) {
#pragma unroll
        for (int ky = 0; ky < 4; ++ky) {
            const float* xr = xb + ic * 400 + (oy * 2 + ky) * 20 + ox * 2;
            const float* wr = w + oc * 512 + ic * 16 + ky * 4;
#pragma unroll
            for (int kx = 0; kx < 4; ++kx) acc += xr[kx] * wr[kx];
        }
    }
    out[idx] = fmaxf(acc + bias[oc], 0.0f);
}

__global__ void conv3_k(const float* __restrict__ h2, const float* __restrict__ w,
                        const float* __restrict__ bias, _Float16* __restrict__ out)
{   // (256,64,9,9) -> (256,64,7,7), k=3 s=1, ReLU, f16 out (feeds fc WMMA)
    int idx = blockIdx.x * blockDim.x + threadIdx.x;
    if (idx >= kB * 64 * 49) return;
    int ox = idx % 7, oy = (idx / 7) % 7, oc = (idx / 49) % 64, b = idx / (49 * 64);
    const float* xb = h2 + (size_t)b * 64 * 81;
    float acc = 0.0f;
    for (int ic = 0; ic < 64; ++ic) {
#pragma unroll
        for (int ky = 0; ky < 3; ++ky) {
            const float* xr = xb + ic * 81 + (oy + ky) * 9 + ox;
            const float* wr = w + oc * 576 + ic * 9 + ky * 3;
#pragma unroll
            for (int kx = 0; kx < 3; ++kx) acc += xr[kx] * wr[kx];
        }
    }
    out[idx] = (_Float16)fmaxf(acc + bias[oc], 0.0f);
}

// ---------------------------------------------------------------------------
// Elementwise / pack kernels
// ---------------------------------------------------------------------------
__global__ void cvt_f32_f16(const float* __restrict__ in,
                            _Float16* __restrict__ out, int n)
{
    int i = blockIdx.x * blockDim.x + threadIdx.x;
    if (i < n) out[i] = (_Float16)in[i];
}

__global__ void pack_seq_k(const float* __restrict__ mem,
                           const float* __restrict__ xenc,
                           _Float16* __restrict__ seq)
{   // seq[b,l,:] = l<64 ? memory_window[b,l,:] : x_enc[b,:]
    int idx = blockIdx.x * blockDim.x + threadIdx.x;
    if (idx >= kROWS * kHID) return;
    int h = idx % kHID, l = (idx / kHID) % kL, b = idx / (kHID * kL);
    float v = (l < kMEM) ? mem[((size_t)b * kMEM + l) * kHID + h]
                         : xenc[(size_t)b * kHID + h];
    seq[idx] = (_Float16)v;
}

__global__ void dwconv_silu_k(const _Float16* __restrict__ xz,
                              const float* __restrict__ w,
                              const float* __restrict__ bias,
                              _Float16* __restrict__ out)
{   // causal depthwise conv (k=4, left pad 3) over L, then SiLU
    int idx = blockIdx.x * blockDim.x + threadIdx.x;
    if (idx >= kROWS * kDIN) return;
    int d = idx % kDIN, l = (idx / kDIN) % kL, b = idx / (kDIN * kL);
    float acc = bias[d];
#pragma unroll
    for (int k = 0; k < 4; ++k) {
        int ls = l - 3 + k;
        if (ls >= 0)
            acc += (float)xz[((size_t)b * kL + ls) * (2 * kDIN) + d] * w[d * 4 + k];
    }
    out[idx] = (_Float16)(acc / (1.0f + __expf(-acc)));
}

__global__ void dtproj_k(const float* __restrict__ xdbl,
                         const float* __restrict__ wt,
                         const float* __restrict__ bias,
                         float* __restrict__ dt)
{   // dt = softplus(xdbl[:, :32] @ wt^T + b)   K=32
    int idx = blockIdx.x * blockDim.x + threadIdx.x;
    if (idx >= kROWS * kDIN) return;
    int d = idx % kDIN, row = idx / kDIN;
    float acc = bias[d];
    const float* xr = xdbl + (size_t)row * 64;
    const float* wr = wt + (size_t)d * 32;
#pragma unroll
    for (int k = 0; k < 32; ++k) acc += xr[k] * wr[k];
    dt[idx] = (acc > 20.0f) ? acc : __logf(1.0f + __expf(acc));
}

__global__ void scan_k(const float* __restrict__ dts,
                       const float* __restrict__ xdbl,
                       const _Float16* __restrict__ xc,
                       const _Float16* __restrict__ xz,
                       const float* __restrict__ alog,
                       const float* __restrict__ Dp,
                       _Float16* __restrict__ y16)
{   // selective scan; only y at l=L-1 needed. grid = (DIN/256, B)
    int b = blockIdx.y;
    int d = blockIdx.x * blockDim.x + threadIdx.x;
    __shared__ float sB[16], sC[16];
    float Ad[16];
#pragma unroll
    for (int s = 0; s < 16; ++s) Ad[s] = -__expf(alog[(size_t)d * 16 + s]);
    float h[16];
#pragma unroll
    for (int s = 0; s < 16; ++s) h[s] = 0.0f;
    float ulast = 0.0f;
    for (int l = 0; l < kL; ++l) {
        size_t row = (size_t)b * kL + l;
        float bc = 0.0f;
        if (threadIdx.x < 32) bc = xdbl[row * 64 + 32 + threadIdx.x];
        __syncthreads();
        if (threadIdx.x < 16)       sB[threadIdx.x] = bc;
        else if (threadIdx.x < 32)  sC[threadIdx.x - 16] = bc;
        __syncthreads();
        float dtv = dts[row * kDIN + d];
        float u   = (float)xc[row * kDIN + d];
        float du  = dtv * u;
#pragma unroll
        for (int s = 0; s < 16; ++s)
            h[s] = __expf(dtv * Ad[s]) * h[s] + du * sB[s];
        ulast = u;
    }
    float y = 0.0f;
#pragma unroll
    for (int s = 0; s < 16; ++s) y += h[s] * sC[s];
    y += ulast * Dp[d];
    float zv = (float)xz[((size_t)b * kL + (kL - 1)) * (2 * kDIN) + kDIN + d];
    float gate = zv / (1.0f + __expf(-zv));          // silu(z)
    y16[(size_t)b * kDIN + d] = (_Float16)(y * gate);
}

__global__ void heads_k(const float* __restrict__ cur,
                        const float* __restrict__ aw, const float* __restrict__ ab,
                        const float* __restrict__ cw, const float* __restrict__ cb,
                        float* __restrict__ logits, float* __restrict__ value)
{
    int idx = blockIdx.x * blockDim.x + threadIdx.x;
    if (idx >= kB * 19) return;
    int b = idx / 19, j = idx % 19;
    const float* c = cur + (size_t)b * kHID;
    if (j < kNACT) {
        float acc = ab[j];
        const float* w = aw + (size_t)j * kHID;
        for (int k = 0; k < kHID; ++k) acc += c[k] * w[k];
        logits[b * kNACT + j] = acc;
    } else {
        float acc = cb[0];
        for (int k = 0; k < kHID; ++k) acc += c[k] * cw[k];
        value[b] = acc;
    }
}

__global__ void memout_k(const float* __restrict__ mem,
                         const float* __restrict__ cur,
                         float* __restrict__ out)
{   // new_memory = concat(memory_window[:,1:], cur)
    int idx = blockIdx.x * blockDim.x + threadIdx.x;
    if (idx >= kB * kMEM * kHID) return;
    int h = idx % kHID, t = (idx / kHID) % kMEM, b = idx / (kHID * kMEM);
    out[idx] = (t < kMEM - 1) ? mem[((size_t)b * kMEM + t + 1) * kHID + h]
                              : cur[(size_t)b * kHID + h];
}

__global__ void copy_k(const float* __restrict__ in, float* __restrict__ out, int n)
{
    int i = blockIdx.x * blockDim.x + threadIdx.x;
    if (i < n) out[i] = in[i];
}

// ---------------------------------------------------------------------------
extern "C" void kernel_launch(void* const* d_in, const int* in_sizes, int n_in,
                              void* d_out, int out_size, void* d_ws, size_t ws_size,
                              hipStream_t stream)
{
    (void)in_sizes; (void)n_in; (void)out_size; (void)ws_size;
    const float* x    = (const float*)d_in[0];
    const float* memw = (const float*)d_in[1];
    const float* c1w  = (const float*)d_in[2];
    const float* c1b  = (const float*)d_in[3];
    const float* c2w  = (const float*)d_in[4];
    const float* c2b  = (const float*)d_in[5];
    const float* c3w  = (const float*)d_in[6];
    const float* c3b  = (const float*)d_in[7];
    const float* fcw  = (const float*)d_in[8];
    const float* fcb  = (const float*)d_in[9];
    const float* ipw  = (const float*)d_in[10];
    const float* c1dw = (const float*)d_in[11];
    const float* c1db = (const float*)d_in[12];
    const float* xpw  = (const float*)d_in[13];
    const float* dtw  = (const float*)d_in[14];
    const float* dtbv = (const float*)d_in[15];
    const float* alog = (const float*)d_in[16];
    const float* Dp   = (const float*)d_in[17];
    const float* opw  = (const float*)d_in[18];
    const float* aw   = (const float*)d_in[19];
    const float* ab   = (const float*)d_in[20];
    const float* cw   = (const float*)d_in[21];
    const float* cb   = (const float*)d_in[22];

    char* ws = (char*)d_ws;
    size_t off = 0;
    auto alloc = [&](size_t bytes) -> void* {
        void* p = ws + off;
        off = (off + bytes + 255) & ~(size_t)255;
        return p;
    };

    float*    h1    = (float*)   alloc((size_t)kB * 32 * 400 * 4);
    float*    h2    = (float*)   alloc((size_t)kB * 64 * 81  * 4);
    _Float16* h3    = (_Float16*)alloc((size_t)kB * kFCK * 2);
    float*    xenc  = (float*)   alloc((size_t)kB * kHID * 4);
    _Float16* seq16 = (_Float16*)alloc((size_t)kROWS * kHID * 2);
    _Float16* fcw16 = (_Float16*)alloc((size_t)kHID * kFCK * 2);
    _Float16* ipw16 = (_Float16*)alloc((size_t)2 * kDIN * kHID * 2);
    _Float16* xpw16 = (_Float16*)alloc((size_t)64 * kDIN * 2);
    _Float16* opw16 = (_Float16*)alloc((size_t)kHID * kDIN * 2);
    _Float16* xz16  = (_Float16*)alloc((size_t)kROWS * 2 * kDIN * 2);
    _Float16* xc16  = (_Float16*)alloc((size_t)kROWS * kDIN * 2);
    float*    xdbl  = (float*)   alloc((size_t)kROWS * 64 * 4);
    float*    dts   = (float*)   alloc((size_t)kROWS * kDIN * 4);
    _Float16* y16   = (_Float16*)alloc((size_t)kB * kDIN * 2);
    float*    cur   = (float*)   alloc((size_t)kB * kHID * 4);

    float* out        = (float*)d_out;
    float* out_logits = out;
    float* out_value  = out + kB * kNACT;
    float* out_mem    = out_value + kB;
    float* out_cur    = out_mem + (size_t)kB * kMEM * kHID;

    const int T = 256;

    // --- encoder ---
    conv1_k<<<(kB * 32 * 400 + T - 1) / T, T, 0, stream>>>(x, c1w, c1b, h1);
    conv2_k<<<(kB * 64 * 81 + T - 1) / T, T, 0, stream>>>(h1, c2w, c2b, h2);
    conv3_k<<<(kB * 64 * 49 + T - 1) / T, T, 0, stream>>>(h2, c3w, c3b, h3);

    // --- weight f32 -> f16 packs ---
    cvt_f32_f16<<<(kHID * kFCK + T - 1) / T, T, 0, stream>>>(fcw, fcw16, kHID * kFCK);
    cvt_f32_f16<<<(2 * kDIN * kHID + T - 1) / T, T, 0, stream>>>(ipw, ipw16, 2 * kDIN * kHID);
    cvt_f32_f16<<<(64 * kDIN + T - 1) / T, T, 0, stream>>>(xpw, xpw16, 64 * kDIN);
    cvt_f32_f16<<<(kHID * kDIN + T - 1) / T, T, 0, stream>>>(opw, opw16, kHID * kDIN);

    // fc: xenc = relu(h3 @ fcw^T + fcb)     M=256 N=512 K=3136
    gemm_wmma_f16<<<dim3(kHID / 128, kB / 64), 128, 0, stream>>>(
        h3, fcw16, fcb, xenc, nullptr, kB, kHID, kFCK, 1);

    // seq = concat(memory_window, x_enc), f16
    pack_seq_k<<<(kROWS * kHID + T - 1) / T, T, 0, stream>>>(memw, xenc, seq16);

    // in_proj: xz = seq @ ipw^T             M=16640 N=2048 K=512 (f16 out)
    gemm_wmma_f16<<<dim3((2 * kDIN) / 128, kROWS / 64), 128, 0, stream>>>(
        seq16, ipw16, nullptr, nullptr, xz16, kROWS, 2 * kDIN, kHID, 0);

    // depthwise causal conv1d + SiLU
    dwconv_silu_k<<<(kROWS * kDIN + T - 1) / T, T, 0, stream>>>(xz16, c1dw, c1db, xc16);

    // x_dbl = xc @ xpw^T                    M=16640 N=64 K=1024
    gemm_wmma_f16<<<dim3(1, kROWS / 64), 128, 0, stream>>>(
        xc16, xpw16, nullptr, xdbl, nullptr, kROWS, 64, kDIN, 0);

    // dt = softplus(x_dbl[:, :32] @ dtw^T + dtb)
    dtproj_k<<<(kROWS * kDIN + T - 1) / T, T, 0, stream>>>(xdbl, dtw, dtbv, dts);

    // selective scan -> gated y at last timestep (f16)
    scan_k<<<dim3(kDIN / 256, kB), 256, 0, stream>>>(dts, xdbl, xc16, xz16, alog, Dp, y16);

    // out_proj: cur = y @ opw^T             M=256 N=512 K=1024
    gemm_wmma_f16<<<dim3(kHID / 128, kB / 64), 128, 0, stream>>>(
        y16, opw16, nullptr, cur, nullptr, kB, kHID, kDIN, 0);

    // heads + outputs
    heads_k<<<(kB * 19 + T - 1) / T, T, 0, stream>>>(cur, aw, ab, cw, cb,
                                                     out_logits, out_value);
    memout_k<<<(kB * kMEM * kHID + T - 1) / T, T, 0, stream>>>(memw, cur, out_mem);
    copy_k<<<(kB * kHID + T - 1) / T, T, 0, stream>>>(cur, out_cur, kB * kHID);
}
